// SRUBaseline_48223892799513
// MI455X (gfx1250) — matrix operand order repaired and checked by reading, not compiled, not run on hardware
//
#include <hip/hip_runtime.h>
#include <hip/hip_bf16.h>
#include <hip/hip_fp16.h>
#include <math.h>

typedef _Float16 half_t;
typedef __attribute__((ext_vector_type(16))) _Float16 v16h;
typedef __attribute__((ext_vector_type(8)))  _Float16 v8h;
typedef __attribute__((ext_vector_type(8)))  float    v8f;

#define BB   32
#define FFd  2048
#define JC   51
#define HH   256
#define H3   768
#define ROWS (BB * FFd)        // 65536 rows (b,f)

// ---------------------------------------------------------------------------
// Pack W_l [L,256,768] fp32 -> f16 WMMA B-fragments.
// Packed layout per layer: tile index t = nt*8 + kt (Kt=8, Nt=48),
// element ((t*32 + lane)*16 + hh)  where lane: n = nt*16+(lane&15),
// hh: k = kt*32 + 16*(lane>>4) + hh   (B 32x16 fragment: halves = consecutive K)
// ---------------------------------------------------------------------------
__global__ __launch_bounds__(256) void pack_wl_kernel(const float* __restrict__ W,
                                                      half_t* __restrict__ P) {
    int idx = blockIdx.x * 256 + threadIdx.x;            // < 2*196608
    int l    = idx / 196608;
    int rem  = idx % 196608;
    int hh   = rem & 15;
    int lane = (rem >> 4) & 31;
    int tile = rem >> 9;
    int kt   = tile & 7;
    int nt   = tile >> 3;
    int g    = lane >> 4;
    int k    = kt * 32 + 16 * g + hh;
    int n    = nt * 16 + (lane & 15);
    P[idx] = (half_t)W[((size_t)(l * HH + k)) * H3 + n];
}

// W_out [256,51] -> padded [256,64] packed fragments (Kt=8, Nt=4)
__global__ __launch_bounds__(256) void pack_wout_kernel(const float* __restrict__ W,
                                                        half_t* __restrict__ P) {
    int idx  = blockIdx.x * 256 + threadIdx.x;           // < 16384
    int hh   = idx & 15;
    int lane = (idx >> 4) & 31;
    int tile = idx >> 9;
    int kt   = tile & 7;
    int nt   = tile >> 3;
    int g    = lane >> 4;
    int k    = kt * 32 + 16 * g + hh;
    int n    = nt * 16 + (lane & 15);
    P[idx] = (n < JC) ? (half_t)W[(size_t)k * JC + n] : (half_t)0.0f;
}

// ---------------------------------------------------------------------------
// Input projection: h[row,h] = x[row,:] . W_in[:,h] + b_in[h]  (K=51, VALU)
// Each block: 64 rows, 256 threads (thread = output column h).
// W_in column held in registers; x row staged through LDS.
// ---------------------------------------------------------------------------
__global__ __launch_bounds__(256) void sru_input_kernel(const float* __restrict__ x,
                                                        const float* __restrict__ Win,
                                                        const float* __restrict__ bin,
                                                        half_t* __restrict__ hA) {
    __shared__ float xs[JC];
    const int h = threadIdx.x;
    float wc[JC];
#pragma unroll
    for (int i = 0; i < JC; ++i) wc[i] = Win[i * HH + h];
    const float bh = bin[h];
    const int row0 = blockIdx.x * 64;
    for (int r = 0; r < 64; ++r) {
        const int row = row0 + r;
        __syncthreads();
        if (threadIdx.x < JC) xs[threadIdx.x] = x[(size_t)row * JC + threadIdx.x];
        __syncthreads();
        float acc = bh;
#pragma unroll
        for (int i = 0; i < JC; ++i) acc += xs[i] * wc[i];
        hA[(size_t)row * HH + h] = (half_t)acc;
    }
}

// ---------------------------------------------------------------------------
// Builds a 16x32 f16 A fragment from a per-lane row pointer (layout:
// lanes0-15 K=0..15, lanes16-31 K=16..31; group shift folded into ap).
// ---------------------------------------------------------------------------
__device__ __forceinline__ v16h load_a_frag(const half_t* ap) {
    v8h alo = *(const v8h*)(ap);
    v8h ahi = *(const v8h*)(ap + 8);
    return __builtin_shufflevector(alo, ahi, 0, 1, 2, 3, 4, 5, 6, 7,
                                   8, 9, 10, 11, 12, 13, 14, 15);
}

// ---------------------------------------------------------------------------
// WMMA GEMM:  Out[M x N] = A[M x 256](f16) * Bpacked + bias, fused epilogue.
// One wave computes a 32(M) x 64(N) strip: 8 accumulators, K=256 fully
// unrolled (8 steps x 8 v_wmma); each B fragment feeds two WMMAs (M reuse).
// MODE 0: layer GEMM (N=768): store fp32 u; sigmoid applied for cols >= 256.
// MODE 1: output GEMM (N=64 padded): store cols < 51 to y.
// ---------------------------------------------------------------------------
template <int MODE>
__global__ __launch_bounds__(256) void sru_gemm_kernel(const half_t* __restrict__ A,
                                                       const half_t* __restrict__ Bp,
                                                       const float* __restrict__ bias,
                                                       float* __restrict__ Out,
                                                       int NG) {
    const int lane = threadIdx.x & 31;
    const int wave = threadIdx.x >> 5;
    const int job  = blockIdx.x * 8 + wave;
    const int mt   = job / NG;              // 32-row M tile
    const int ng   = job % NG;              // group of 4 N tiles (64 cols)
    const int g    = lane >> 4;
    const int nl   = lane & 15;

    // Per-lane base pointers (A group shift folded in)
    const half_t* ap0 = A + (size_t)(mt * 32 + nl) * HH + 16 * g;
    const half_t* ap1 = ap0 + (size_t)16 * HH;
    const half_t* bp  = Bp + ((size_t)(ng * 4) * 8 * 32 + lane) * 16;

    v8f acc[8];
#pragma unroll
    for (int j = 0; j < 8; ++j) acc[j] = (v8f){0.f, 0.f, 0.f, 0.f, 0.f, 0.f, 0.f, 0.f};

#pragma unroll
    for (int kt = 0; kt < 8; ++kt) {
        const int off = kt * 32;
        v16h a0 = load_a_frag(ap0 + off);
        v16h a1 = load_a_frag(ap1 + off);
#pragma unroll
        for (int j = 0; j < 4; ++j) {
            const v16h b = *(const v16h*)(bp + (size_t)j * 8 * 512 + kt * 512);
            acc[2 * j]     = __builtin_amdgcn_wmma_f32_16x16x32_f16(
                false, a0, false, b, (short)0, acc[2 * j], false, false);
            acc[2 * j + 1] = __builtin_amdgcn_wmma_f32_16x16x32_f16(
                false, a1, false, b, (short)0, acc[2 * j + 1], false, false);
        }
    }

    // Epilogue: C/D layout: lane n = lane&15, VGPR r -> row m = 8*(lane>>4)+r
#pragma unroll
    for (int j = 0; j < 4; ++j) {
        const int col = (ng * 4 + j) * 16 + nl;
        float bv;
        if (MODE == 0) bv = bias[col];
        else           bv = (col < JC) ? bias[col] : 0.0f;
#pragma unroll
        for (int i = 0; i < 2; ++i) {
            union { v8f v; float f[8]; } ua;
            ua.v = acc[2 * j + i];
#pragma unroll
            for (int r = 0; r < 8; ++r) {
                const int row = mt * 32 + 16 * i + 8 * g + r;
                float v = ua.f[r] + bv;
                if (MODE == 0) {
                    if (col >= HH) v = 1.0f / (1.0f + __expf(-v));  // f and r gates
                    Out[(size_t)row * H3 + col] = v;
                } else {
                    if (col < JC) Out[(size_t)row * JC + col] = v;
                }
            }
        }
    }
}

// ---------------------------------------------------------------------------
// SRU recurrence + output mix. One thread per (b,h): 8192 threads.
// c_t = f*c + (1-f)*x~ ; h = r*tanh(c) + (1-r)*x~ ; write next-layer A (f16).
// Loads are address-independent of the recurrence, so the compiler can keep
// many iterations of loads in flight; only the FMA chain is serial.
// ---------------------------------------------------------------------------
__global__ __launch_bounds__(256) void sru_scan_kernel(const float* __restrict__ u,
                                                       half_t* __restrict__ hA) {
    const int gid = blockIdx.x * 256 + threadIdx.x;   // 0..8191
    const int b = gid >> 8;
    const int h = gid & 255;
    float c = 0.0f;
    const size_t base  = (size_t)b * FFd * H3 + h;
    const size_t obase = (size_t)b * FFd * HH + h;
    for (int f = 0; f < FFd; ++f) {
        const size_t p = base + (size_t)f * H3;
        const float xt = u[p];
        const float ft = u[p + HH];         // already sigmoid'ed in GEMM epilogue
        const float rt = u[p + 2 * HH];
        c = ft * c + (1.0f - ft) * xt;
        const float hv = rt * tanhf(c) + (1.0f - rt) * xt;
        hA[obase + (size_t)f * HH] = (half_t)hv;
    }
}

// ---------------------------------------------------------------------------
extern "C" void kernel_launch(void* const* d_in, const int* in_sizes, int n_in,
                              void* d_out, int out_size, void* d_ws, size_t ws_size,
                              hipStream_t stream) {
    const float* x    = (const float*)d_in[0];
    const float* Win  = (const float*)d_in[1];
    const float* bin  = (const float*)d_in[2];
    const float* Wl   = (const float*)d_in[3];
    const float* bl   = (const float*)d_in[4];
    const float* Wout = (const float*)d_in[5];
    const float* bout = (const float*)d_in[6];
    float* out = (float*)d_out;

    char* ws = (char*)d_ws;
    half_t* hA  = (half_t*)(ws);                                      // 33,554,432 B
    float*  uB  = (float*)(ws + 33554432ull);                         // 201,326,592 B
    half_t* WlP = (half_t*)(ws + 33554432ull + 201326592ull);         // 786,432 B
    half_t* WoP = (half_t*)(ws + 33554432ull + 201326592ull + 786432ull); // 32,768 B

    // Weight packing (tiny, f32 -> f16 fragment layout)
    pack_wl_kernel<<<1536, 256, 0, stream>>>(Wl, WlP);
    pack_wout_kernel<<<64, 256, 0, stream>>>(Wout, WoP);

    // Input projection -> hA (f16)
    sru_input_kernel<<<1024, 256, 0, stream>>>(x, Win, bin, hA);

    // Two SRU layers: WMMA GEMM (u = h W + b, gates sigmoided) then scan
    for (int l = 0; l < 2; ++l) {
        // jobs = 2048 M-tiles * 12 N-groups = 24576 waves -> 3072 blocks
        sru_gemm_kernel<0><<<3072, 256, 0, stream>>>(hA, WlP + (size_t)l * 196608,
                                                     bl + l * H3, uB, 12);
        sru_scan_kernel<<<32, 256, 0, stream>>>(uB, hA);
    }

    // Output projection (N padded to 64, only cols < 51 written)
    // jobs = 2048 M-tiles -> 256 blocks
    sru_gemm_kernel<1><<<256, 256, 0, stream>>>(hA, WoP, bout, out, 1);
}
